// CG_57638461112745
// MI455X (gfx1250) — matrix operand order repaired and checked by reading, not compiled, not run on hardware
//
#include <hip/hip_runtime.h>
#include <hip/hip_bf16.h>

typedef __attribute__((ext_vector_type(16))) _Float16 v16h;
typedef __attribute__((ext_vector_type(8)))  float    v8f;

#define IN_F   128
#define HID_F  256
#define OUT_F  128
#define EPS    1e-5f

// ---------------------------------------------------------------------------
// Small utility kernels
// ---------------------------------------------------------------------------
__global__ void fill_kernel(float* __restrict__ p, float v, int n) {
    int i = blockIdx.x * blockDim.x + threadIdx.x;
    if (i < n) p[i] = v;
}

__global__ void copy_kernel(float* __restrict__ dst, const float* __restrict__ src, int n) {
    int i = blockIdx.x * blockDim.x + threadIdx.x;
    if (i < n) dst[i] = src[i];
}

__global__ void deg_edges_kernel(const int* __restrict__ dst, float* __restrict__ deg, int E) {
    int e = blockIdx.x * blockDim.x + threadIdx.x;
    if (e < E) atomicAdd(&deg[dst[e]], 1.0f);
}

__global__ void rsqrt_inplace_kernel(float* __restrict__ p, int n) {
    int i = blockIdx.x * blockDim.x + threadIdx.x;
    if (i < n) p[i] = rsqrtf(p[i]);
}

__global__ void mask_set_kernel(float* __restrict__ h, const int* __restrict__ mask,
                                const float* __restrict__ token, int nmask, int F) {
    int i = blockIdx.x * blockDim.x + threadIdx.x;
    if (i >= nmask * F) return;
    int r = mask[i / F], f = i % F;
    h[(size_t)r * F + f] = token[f];
}

__global__ void mask_zero_kernel(float* __restrict__ h, const int* __restrict__ mask,
                                 int nmask, int F) {
    int i = blockIdx.x * blockDim.x + threadIdx.x;
    if (i >= nmask * F) return;
    int r = mask[i / F], f = i % F;
    h[(size_t)r * F + f] = 0.0f;
}

// ---------------------------------------------------------------------------
// Pre-pack fp32 weight W[K,N] into f16 in WMMA B-fragment register order.
// Fragment (nt, ks): 32 lanes x 16 halves, element j of lane l holds
//   B[k, col],  k = ks*32 + (j<8 ? (l>>4)*8 + j : 16 + (l>>4)*8 + j-8),  col = nt*16 + (l&15)
// stored contiguously so the GEMM reads one aligned 32B vector per fragment.
// ---------------------------------------------------------------------------
__global__ void pack_w_kernel(const float* __restrict__ W, _Float16* __restrict__ Wp,
                              int K, int N) {
    const int ksteps = K >> 5;
    const int total  = (N >> 4) * ksteps * 512;
    int i = blockIdx.x * blockDim.x + threadIdx.x;
    if (i >= total) return;
    int j    = i & 15;
    int lane = (i >> 4) & 31;
    int ks   = (i >> 9) % ksteps;
    int nt   = (i >> 9) / ksteps;
    int k    = ks * 32 + (j < 8 ? (lane >> 4) * 8 + j : 16 + (lane >> 4) * 8 + (j - 8));
    int col  = nt * 16 + (lane & 15);
    Wp[i] = (_Float16)W[(size_t)k * N + col];
}

// ---------------------------------------------------------------------------
// WMMA f16 GEMM: C[M,N] = A[M,K] * Bp (pre-packed f16 fragments).
// One wave computes a 16x64 tile (4 N-tiles, 4 accumulators) -> A fragment
// reused 4x per K-step. K is compile-time so the loop fully unrolls.
// Block = 256 threads = 8 waves = 8 M-tiles. grid.y covers N/64.
// ---------------------------------------------------------------------------
template <int K>
__global__ void gemm_wmma_kernel(const float* __restrict__ A,
                                 const _Float16* __restrict__ Bp,
                                 float* __restrict__ C,
                                 int M, int N) {
    constexpr int KSTEPS = K / 32;
    const int lane = threadIdx.x & 31;
    const int wave = threadIdx.x >> 5;
    const int half = lane >> 4;         // 0: lanes 0-15, 1: lanes 16-31
    const int l16  = lane & 15;

    const int mtiles = M >> 4;
    const int mtile  = blockIdx.x * 8 + wave;   // wave-uniform
    if (mtile >= mtiles) return;                // uniform branch: EXEC all-ones
    const int ntb  = blockIdx.y * 4;            // first of 4 N-tiles
    const int row0 = mtile * 16;

    const float*    aRow = A + (size_t)(row0 + l16) * K;
    const _Float16* bPtr = Bp + ((size_t)ntb * KSTEPS * 512) + lane * 16;

    v8f acc0 = {}, acc1 = {}, acc2 = {}, acc3 = {};
    #pragma unroll
    for (int ks = 0; ks < KSTEPS; ++ks) {
        const int k0 = ks * 32;
        if (ks + 1 < KSTEPS) __builtin_prefetch(aRow + k0 + 32, 0, 1);

        // A fragment, 16x32 f16 (ISA 7.12.2): lanes 0-15 K {k0..k0+7, k0+16..23},
        // lanes 16-31 shifted by 8. fp32 loads + cvt (fused conversion).
        const int b1 = k0 + half * 8;
        const int b2 = k0 + 16 + half * 8;
        float4 f0 = *(const float4*)(aRow + b1);
        float4 f1 = *(const float4*)(aRow + b1 + 4);
        float4 f2 = *(const float4*)(aRow + b2);
        float4 f3 = *(const float4*)(aRow + b2 + 4);
        v16h a;
        a[0]=(_Float16)f0.x;  a[1]=(_Float16)f0.y;  a[2]=(_Float16)f0.z;  a[3]=(_Float16)f0.w;
        a[4]=(_Float16)f1.x;  a[5]=(_Float16)f1.y;  a[6]=(_Float16)f1.z;  a[7]=(_Float16)f1.w;
        a[8]=(_Float16)f2.x;  a[9]=(_Float16)f2.y;  a[10]=(_Float16)f2.z; a[11]=(_Float16)f2.w;
        a[12]=(_Float16)f3.x; a[13]=(_Float16)f3.y; a[14]=(_Float16)f3.z; a[15]=(_Float16)f3.w;

        // 4 pre-packed B fragments: one aligned 32B vector load each
        const _Float16* bp = bPtr + (size_t)ks * 512;
        v16h b0 = *(const v16h*)(bp);
        v16h b1v = *(const v16h*)(bp + (size_t)KSTEPS * 512);
        v16h b2v = *(const v16h*)(bp + (size_t)2 * KSTEPS * 512);
        v16h b3v = *(const v16h*)(bp + (size_t)3 * KSTEPS * 512);

        acc0 = __builtin_amdgcn_wmma_f32_16x16x32_f16(false, a, false, b0,  (short)0, acc0, false, false);
        acc1 = __builtin_amdgcn_wmma_f32_16x16x32_f16(false, a, false, b1v, (short)0, acc1, false, false);
        acc2 = __builtin_amdgcn_wmma_f32_16x16x32_f16(false, a, false, b2v, (short)0, acc2, false, false);
        acc3 = __builtin_amdgcn_wmma_f32_16x16x32_f16(false, a, false, b3v, (short)0, acc3, false, false);
    }

    // C store: VGPR i -> M = row0 + half*8 + i, N = (ntb+nt)*16 + l16
    #pragma unroll
    for (int nt = 0; nt < 4; ++nt) {
        const v8f acc = nt == 0 ? acc0 : nt == 1 ? acc1 : nt == 2 ? acc2 : acc3;
        const int col = (ntb + nt) * 16 + l16;
        #pragma unroll
        for (int i = 0; i < 8; ++i)
            C[(size_t)(row0 + half * 8 + i) * N + col] = acc[i];
    }
}

// ---------------------------------------------------------------------------
// agg = hw * dinv^2 (self-loop) + bias
// ---------------------------------------------------------------------------
__global__ void agg_init_kernel(const float* __restrict__ hw, const float* __restrict__ dinv,
                                const float* __restrict__ bias, float* __restrict__ agg,
                                int F, int Nn) {
    int i = blockIdx.x * blockDim.x + threadIdx.x;
    if (i >= Nn * F) return;
    int n = i / F, f = i % F;
    float di = dinv[n];
    agg[i] = hw[i] * di * di + bias[f];
}

// ---------------------------------------------------------------------------
// Edge scatter: one wave per edge, wave-uniform src/dst/coef, float4 gathers,
// f32 global atomic adds. Per-layer feature matrices (<=51MB) are L2-resident
// (192MB), so gathers/atomics stay on-chip.
// ---------------------------------------------------------------------------
__global__ void agg_edges_kernel(const int* __restrict__ src, const int* __restrict__ dst,
                                 const float* __restrict__ hw, const float* __restrict__ dinv,
                                 float* __restrict__ agg, int F, int E) {
    int wave = (blockIdx.x * blockDim.x + threadIdx.x) >> 5;
    int lane = threadIdx.x & 31;
    if (wave >= E) return;
    int s = src[wave], d = dst[wave];
    float coef = dinv[s] * dinv[d];
    const float4* hs = (const float4*)(hw + (size_t)s * F);
    float* ad = agg + (size_t)d * F;
    for (int q = lane; q * 4 < F; q += 32) {
        float4 v = hs[q];
        atomicAdd(ad + 4 * q + 0, v.x * coef);
        atomicAdd(ad + 4 * q + 1, v.y * coef);
        atomicAdd(ad + 4 * q + 2, v.z * coef);
        atomicAdd(ad + 4 * q + 3, v.w * coef);
    }
}

// ---------------------------------------------------------------------------
// BatchNorm: column partial sums (blockDim == F), then apply + PReLU
// ---------------------------------------------------------------------------
#define BN_ROWS 256
__global__ void bn_partial_kernel(const float* __restrict__ h, float* __restrict__ sums,
                                  int F, int rows) {
    int c  = threadIdx.x;               // blockDim.x == F
    int r0 = blockIdx.x * BN_ROWS;
    int r1 = r0 + BN_ROWS; if (r1 > rows) r1 = rows;
    float s = 0.f, sq = 0.f;
    for (int r = r0; r < r1; ++r) {
        float v = h[(size_t)r * F + c];
        s += v; sq += v * v;
    }
    atomicAdd(&sums[c], s);
    atomicAdd(&sums[F + c], sq);
}

__global__ void bn_apply_kernel(float* __restrict__ h, const float* __restrict__ sums,
                                const float* __restrict__ g, const float* __restrict__ be,
                                const float* __restrict__ a, int F, int rows) {
    int i = blockIdx.x * blockDim.x + threadIdx.x;
    if (i >= rows * F) return;
    int c = i % F;
    float inv = 1.0f / (float)rows;
    float mu  = sums[c] * inv;
    float var = sums[F + c] * inv - mu * mu;
    if (var < 0.f) var = 0.f;
    float v = (h[i] - mu) * rsqrtf(var + EPS) * g[c] + be[c];
    h[i] = (v >= 0.f) ? v : a[0] * v;
}

// ---------------------------------------------------------------------------
// SCE loss: one wave per masked row (F == 128 -> 4 floats/lane)
// ---------------------------------------------------------------------------
__global__ void loss_kernel(const float* __restrict__ rex, const float* __restrict__ x,
                            const int* __restrict__ mask, float* __restrict__ out,
                            int nmask) {
    int wave = (blockIdx.x * blockDim.x + threadIdx.x) >> 5;
    int lane = threadIdx.x & 31;
    if (wave >= nmask) return;
    int r = mask[wave];
    const float4 p = *(const float4*)(rex + (size_t)r * IN_F + lane * 4);
    const float4 t = *(const float4*)(x   + (size_t)r * IN_F + lane * 4);
    float pp = p.x*p.x + p.y*p.y + p.z*p.z + p.w*p.w;
    float tt = t.x*t.x + t.y*t.y + t.z*t.z + t.w*t.w;
    float pt = p.x*t.x + p.y*t.y + p.z*t.z + p.w*t.w;
    #pragma unroll
    for (int off = 16; off > 0; off >>= 1) {
        pp += __shfl_xor(pp, off, 32);
        tt += __shfl_xor(tt, off, 32);
        pt += __shfl_xor(pt, off, 32);
    }
    if (lane == 0) {
        float np = sqrtf(pp); if (np < 1e-12f) np = 1e-12f;
        float nt = sqrtf(tt); if (nt < 1e-12f) nt = 1e-12f;
        atomicAdd(out, (1.0f - pt / (np * nt)) / (float)nmask);
    }
}

// ---------------------------------------------------------------------------
// Host orchestration
// ---------------------------------------------------------------------------
extern "C" void kernel_launch(void* const* d_in, const int* in_sizes, int n_in,
                              void* d_out, int out_size, void* d_ws, size_t ws_size,
                              hipStream_t stream) {
    const float* x     = (const float*)d_in[0];
    const int*   ei    = (const int*)  d_in[1];
    const int*   mask  = (const int*)  d_in[2];
    const float* token = (const float*)d_in[3];
    const float* W1 = (const float*)d_in[4];  const float* b1 = (const float*)d_in[5];
    const float* g1 = (const float*)d_in[6];  const float* be1= (const float*)d_in[7];
    const float* a1 = (const float*)d_in[8];
    const float* W2 = (const float*)d_in[9];  const float* b2 = (const float*)d_in[10];
    const float* g2 = (const float*)d_in[11]; const float* be2= (const float*)d_in[12];
    const float* a2 = (const float*)d_in[13];
    const float* Wd = (const float*)d_in[14]; const float* bd = (const float*)d_in[15];
    const float* gd = (const float*)d_in[16]; const float* bed= (const float*)d_in[17];
    const float* ad = (const float*)d_in[18];

    const int N     = in_sizes[0] / IN_F;
    const int E     = in_sizes[1] / 2;
    const int NMASK = in_sizes[2];
    const int* src  = ei;
    const int* dst  = ei + E;

    // Workspace layout (floats)
    float* ws = (float*)d_ws;
    float* D  = ws;                              // N            degrees -> dinv
    float* S  = ws + N;                          // 512          BN sums/sumsq
    float* B1 = S + 512;                         // N*128
    float* B2 = B1 + (size_t)N * 128;            // N*256
    float* B3 = B2 + (size_t)N * 256;            // N*256
    _Float16* W1p = (_Float16*)(B3 + (size_t)N * 256);  // IN*HID f16
    _Float16* W2p = W1p + IN_F * HID_F;                 // HID*OUT f16
    _Float16* Wdp = W2p + HID_F * OUT_F;                // OUT*IN  f16
    float* out = (float*)d_out;

    const int T = 256;
    auto cdiv = [](long a, long b) { return (int)((a + b - 1) / b); };
    const int mblk  = cdiv(N / 16, 8);           // GEMM M-tile blocks
    const int eblk  = cdiv((long)E * 32, T);     // one wave per edge
    const int bnblk = cdiv(N, BN_ROWS);

    // degrees + dinv (deg starts at 1.0 for self-loop)
    fill_kernel<<<cdiv(N, T), T, 0, stream>>>(D, 1.0f, N);
    deg_edges_kernel<<<cdiv(E, T), T, 0, stream>>>(dst, D, E);
    rsqrt_inplace_kernel<<<cdiv(N, T), T, 0, stream>>>(D, N);

    // pre-pack the three weight matrices into f16 fragment order
    pack_w_kernel<<<cdiv(IN_F  * HID_F, T), T, 0, stream>>>(W1, W1p, IN_F,  HID_F);
    pack_w_kernel<<<cdiv(HID_F * OUT_F, T), T, 0, stream>>>(W2, W2p, HID_F, OUT_F);
    pack_w_kernel<<<cdiv(OUT_F * IN_F,  T), T, 0, stream>>>(Wd, Wdp, OUT_F, IN_F);

    // masked input -> B1
    copy_kernel<<<cdiv((long)N * IN_F, T), T, 0, stream>>>(B1, x, N * IN_F);
    mask_set_kernel<<<cdiv((long)NMASK * IN_F, T), T, 0, stream>>>(B1, mask, token, NMASK, IN_F);

    // ---- layer 1: GCN(IN->HID) + BN + PReLU ----
    gemm_wmma_kernel<IN_F><<<dim3(mblk, HID_F / 64), T, 0, stream>>>(B1, W1p, B2, N, HID_F);
    agg_init_kernel<<<cdiv((long)N * HID_F, T), T, 0, stream>>>(B2, D, b1, B3, HID_F, N);
    agg_edges_kernel<<<eblk, T, 0, stream>>>(src, dst, B2, D, B3, HID_F, E);
    fill_kernel<<<2, T, 0, stream>>>(S, 0.0f, 2 * HID_F);
    bn_partial_kernel<<<bnblk, HID_F, 0, stream>>>(B3, S, HID_F, N);
    bn_apply_kernel<<<cdiv((long)N * HID_F, T), T, 0, stream>>>(B3, S, g1, be1, a1, HID_F, N);

    // ---- layer 2: GCN(HID->OUT) + BN + PReLU ----
    gemm_wmma_kernel<HID_F><<<dim3(mblk, OUT_F / 64), T, 0, stream>>>(B3, W2p, B2, N, OUT_F);
    agg_init_kernel<<<cdiv((long)N * OUT_F, T), T, 0, stream>>>(B2, D, b2, B1, OUT_F, N);
    agg_edges_kernel<<<eblk, T, 0, stream>>>(src, dst, B2, D, B1, OUT_F, E);
    fill_kernel<<<1, T, 0, stream>>>(S, 0.0f, 2 * OUT_F);
    bn_partial_kernel<<<bnblk, OUT_F, 0, stream>>>(B1, S, OUT_F, N);
    bn_apply_kernel<<<cdiv((long)N * OUT_F, T), T, 0, stream>>>(B1, S, g2, be2, a2, OUT_F, N);

    // re-mask encoded representation
    mask_zero_kernel<<<cdiv((long)NMASK * OUT_F, T), T, 0, stream>>>(B1, mask, NMASK, OUT_F);

    // ---- decoder: GCN(OUT->IN) + BN + PReLU ----
    gemm_wmma_kernel<OUT_F><<<dim3(mblk, IN_F / 64), T, 0, stream>>>(B1, Wdp, B2, N, IN_F);
    agg_init_kernel<<<cdiv((long)N * IN_F, T), T, 0, stream>>>(B2, D, bd, B3, IN_F, N);
    agg_edges_kernel<<<eblk, T, 0, stream>>>(src, dst, B2, D, B3, IN_F, E);
    fill_kernel<<<1, T, 0, stream>>>(S, 0.0f, 2 * IN_F);
    bn_partial_kernel<<<bnblk, IN_F, 0, stream>>>(B3, S, IN_F, N);
    bn_apply_kernel<<<cdiv((long)N * IN_F, T), T, 0, stream>>>(B3, S, gd, bed, ad, IN_F, N);

    // ---- SCE loss over masked rows ----
    fill_kernel<<<1, 1, 0, stream>>>(out, 0.0f, 1);
    loss_kernel<<<cdiv((long)NMASK * 32, T), T, 0, stream>>>(B3, x, mask, out, NMASK);
}
